// FilterDetections_79937931313581
// MI455X (gfx1250) — 8-bit matrix operand order repaired and checked by reading, compiled, not run on hardware
//
#include <hip/hip_runtime.h>
#include <stdint.h>

#define BATCH 8
#define NBOX  50000
#define NCLS  8
#define K_PRE 1024
#define CAND_CAP 2048
#define MAX_DET 100
#define SCORE_THR 0.99f

#define TRY_ASYNC 1

typedef __attribute__((ext_vector_type(8))) int v8i;

__device__ __forceinline__ bool iou_gt(const float4 a, float aa,
                                       const float4 b, float ab) {
  float iw = fminf(a.z, b.z) - fmaxf(a.x, b.x);
  float ih = fminf(a.w, b.w) - fmaxf(a.y, b.y);
  iw = fmaxf(iw, 0.0f);
  ih = fmaxf(ih, 0.0f);
  float inter = iw * ih;
  return 3.0f * inter > (aa + ab);
}

__global__ void fd_zero_counts(int* gcount) {
  if (threadIdx.x < BATCH) gcount[threadIdx.x] = 0;
}

__global__ __launch_bounds__(512) void fd_nms_per_class(
    const float* __restrict__ boxes, const float* __restrict__ cls,
    uint4* __restrict__ recs, int* __restrict__ gcount, int cap) {
  __shared__ unsigned long long keys[CAND_CAP];
  __shared__ float4 cand[K_PRE];
  __shared__ float area[K_PRE];
  __shared__ unsigned cidx[K_PRE];
  __shared__ unsigned char keep8[K_PRE] __attribute__((aligned(4)));
  __shared__ int pre[16];
  __shared__ int s_cnt;

  const int tid  = threadIdx.x;
  const int nthr = 512;
  const int b = blockIdx.x / NCLS;
  const int c = blockIdx.x % NCLS;

  for (int i = tid; i < CAND_CAP; i += nthr) keys[i] = 0ull;
  for (int i = tid; i < K_PRE; i += nthr) keep8[i] = 0;
  if (tid == 0) s_cnt = 0;
  __syncthreads();

  const float* colp = cls + (size_t)b * NBOX * NCLS + c;
  for (int n = tid; n < NBOX; n += nthr) {
    float s = colp[(size_t)n * NCLS];
    if (s > SCORE_THR) {
      int p = atomicAdd(&s_cnt, 1);
      if (p < CAND_CAP)
        keys[p] = ((unsigned long long)__float_as_uint(s) << 32) |
                  (unsigned)(~(unsigned)n);
    }
  }
  __syncthreads();
  int total = s_cnt;
  if (total > CAND_CAP) total = CAND_CAP;

  for (int k = 2; k <= CAND_CAP; k <<= 1) {
    for (int j = k >> 1; j > 0; j >>= 1) {
      for (int i = tid; i < CAND_CAP; i += nthr) {
        int ixj = i ^ j;
        if (ixj > i) {
          unsigned long long a = keys[i], bb = keys[ixj];
          bool up = (i & k) == 0;
          if ((up && a < bb) || (!up && a > bb)) { keys[i] = bb; keys[ixj] = a; }
        }
      }
      __syncthreads();
    }
  }
  const int V = total < K_PRE ? total : K_PRE;

  for (int i = tid; i < K_PRE; i += nthr) {
    if (i < V) {
      unsigned long long key = keys[i];
      unsigned idx = ~(unsigned)(key & 0xffffffffull);
      cidx[i] = idx;
      unsigned long long ga =
          (unsigned long long)(boxes + ((size_t)b * NBOX + idx) * 4);
#if TRY_ASYNC
      unsigned lds_addr = (unsigned)(uintptr_t)&cand[i];
      asm volatile("global_load_async_to_lds_b128 %0, %1, off"
                   :: "v"(lds_addr), "v"(ga) : "memory");
#else
      cand[i] = *(const float4*)(uintptr_t)ga;
#endif
    } else {
      cand[i] = make_float4(0.f, 0.f, 0.f, 0.f);
      area[i] = 0.0f;
      cidx[i] = 0u;
    }
  }
#if TRY_ASYNC
  asm volatile("s_wait_asynccnt 0x0" ::: "memory");
#endif
  __syncthreads();
  for (int i = tid; i < V; i += nthr) {
    float4 bx = cand[i];
    area[i] = (bx.z - bx.x) * (bx.w - bx.y);
  }
  __syncthreads();

  const int T = (V + 15) >> 4;
  const int wave = tid >> 5;
  const int lane = tid & 31;

  for (int t = 0; t < T; ++t) {
    if (tid < 16) pre[tid] = 0;
    __syncthreads();
    const int i0 = t << 4;
    const int nchunk = (16 * t + 63) >> 6;
    if (wave < nchunk) {
      const int jbase = wave << 6;
      v8i Am;
      {
        const int kb0[8] = {0, 4, 16, 20, 32, 36, 48, 52};
        const int hiA = (lane >= 16) ? 8 : 0;
#pragma unroll
        for (int v = 0; v < 8; ++v)
          Am[v] = *(const int*)((const char*)keep8 + jbase + kb0[v] + hiA);
      }
      const int n = lane & 15;
      const int khi = (lane >> 4) << 4;
      const float4 bi = cand[i0 + n];
      const float ai = area[i0 + n];
      v8i Bm;
#pragma unroll
      for (int v = 0; v < 8; ++v) {
        const int kb = ((v >> 1) << 4) + (khi >> 1) + ((v & 1) << 2);
        unsigned w = 0;
#pragma unroll
        for (int by = 0; by < 4; ++by) {
          const int j = jbase + kb + by;
          if (iou_gt(cand[j], area[j], bi, ai)) w |= 1u << (8 * by);
        }
        Bm[v] = (int)w;
      }
      v8i Cm = (v8i)0;
      v8i Dm = __builtin_amdgcn_wmma_i32_16x16x64_iu8(false, Am, false, Bm, Cm,
                                                      false, false);
      if (lane < 16) atomicAdd(&pre[lane], Dm[0]);
    }
    __syncthreads();
    if (wave == 0) {
      unsigned colmask = 0;
      if (lane < 16) {
        const float4 bi = cand[i0 + lane];
        const float ai = area[i0 + lane];
#pragma unroll
        for (int r = 0; r < 16; ++r) {
          if (r != lane && iou_gt(cand[i0 + r], area[i0 + r], bi, ai))
            colmask |= 1u << r;
        }
      }
      int prv = (lane < 16) ? pre[lane] : 0;
      unsigned keepbits = 0;
      for (int s = 0; s < 16; ++s) {
        unsigned cm = (unsigned)__shfl((int)colmask, s, 32);
        int p = __shfl(prv, s, 32);
        bool sup = (p > 0) || ((cm & keepbits) != 0u);
        if (!sup && (i0 + s) < V) keepbits |= 1u << s;
      }
      if (lane < 16) keep8[i0 + lane] = (unsigned char)((keepbits >> lane) & 1u);
    }
    __syncthreads();
  }

  for (int i = tid; i < V; i += nthr) {
    if (keep8[i]) {
      int p = atomicAdd(&gcount[b], 1);
      if (p < cap) {
        unsigned sb = (unsigned)(keys[i] >> 32);
        recs[(size_t)b * cap + p] =
            make_uint4(sb, cidx[i], (unsigned)c, (unsigned)(c * K_PRE + i));
      }
    }
  }
}

__global__ __launch_bounds__(256) void fd_finalize(
    const float* __restrict__ boxes, const float* __restrict__ rot,
    const float* __restrict__ trans, const uint4* __restrict__ recs,
    const int* __restrict__ gcount, int cap, float* __restrict__ out) {
  const int b = blockIdx.x;
  int m = gcount[b];
  if (m > cap) m = cap;
  const uint4* r = recs + (size_t)b * cap;

  float* ob   = out;
  float* os   = out + BATCH * MAX_DET * 4;
  float* ol   = os + BATCH * MAX_DET;
  float* orot = ol + BATCH * MAX_DET;
  float* otr  = orot + BATCH * MAX_DET * 3;

  for (int s = threadIdx.x; s < MAX_DET; s += blockDim.x) {
    int o = b * MAX_DET + s;
    ob[o * 4 + 0] = ob[o * 4 + 1] = ob[o * 4 + 2] = ob[o * 4 + 3] = -1.0f;
    os[o] = -1.0f;
    ol[o] = -1.0f;
    orot[o * 3 + 0] = orot[o * 3 + 1] = orot[o * 3 + 2] = -1.0f;
    otr[o * 3 + 0] = otr[o * 3 + 1] = otr[o * 3 + 2] = -1.0f;
  }
  __syncthreads();

  for (int e = threadIdx.x; e < m; e += blockDim.x) {
    uint4 me = r[e];
    int rank = 0;
    for (int j = 0; j < m; ++j) {
      uint4 oj = r[j];
      rank += (oj.x > me.x) || (oj.x == me.x && oj.w < me.w);
    }
    if (rank < MAX_DET) {
      int o = b * MAX_DET + rank;
      unsigned idx = me.y;
      const float* gb = boxes + ((size_t)b * NBOX + idx) * 4;
      ob[o * 4 + 0] = gb[0]; ob[o * 4 + 1] = gb[1];
      ob[o * 4 + 2] = gb[2]; ob[o * 4 + 3] = gb[3];
      os[o] = __uint_as_float(me.x);
      ol[o] = (float)(int)me.z;
      const float* gr = rot + ((size_t)b * NBOX + idx) * 3;
      orot[o * 3 + 0] = gr[0]; orot[o * 3 + 1] = gr[1]; orot[o * 3 + 2] = gr[2];
      const float* gt = trans + ((size_t)b * NBOX + idx) * 3;
      otr[o * 3 + 0] = gt[0]; otr[o * 3 + 1] = gt[1]; otr[o * 3 + 2] = gt[2];
    }
  }
}

extern "C" void kernel_launch(void* const* d_in, const int* in_sizes, int n_in,
                              void* d_out, int out_size, void* d_ws,
                              size_t ws_size, hipStream_t stream) {
  const float* boxes = (const float*)d_in[0];
  const float* cls   = (const float*)d_in[1];
  const float* rot   = (const float*)d_in[2];
  const float* trans = (const float*)d_in[3];
  float* out = (float*)d_out;

  int* gcount = (int*)d_ws;
  uint4* recs = (uint4*)((char*)d_ws + 64);
  size_t cap = 8192;
  size_t need = 64 + (size_t)BATCH * cap * sizeof(uint4);
  if (ws_size < need) cap = ws_size > 64 ? (ws_size - 64) / (BATCH * sizeof(uint4)) : 0;

  fd_zero_counts<<<1, 32, 0, stream>>>(gcount);
  fd_nms_per_class<<<BATCH * NCLS, 512, 0, stream>>>(boxes, cls, recs, gcount,
                                                     (int)cap);
  fd_finalize<<<BATCH, 256, 0, stream>>>(boxes, rot, trans, recs, gcount,
                                         (int)cap, out);
}
